// SimpleGravNetModel_69234872811965
// MI455X (gfx1250) — matrix operand order repaired and checked by reading, not compiled, not run on hardware
//
#include <hip/hip_runtime.h>

// Problem constants (match reference)
#define B_SEG  8
#define N_SEG  2048
#define BN     (B_SEG * N_SEG)     // 16384
#define IN_DIM 64
#define PROP   64
#define S_DIM  4
#define KNN    40
#define CAT    (IN_DIM + 2 * PROP) // 192

typedef __attribute__((ext_vector_type(16))) _Float16 v16h;
typedef __attribute__((ext_vector_type(8)))  float    v8f;
typedef __attribute__((ext_vector_type(4)))  unsigned u32x4;
typedef __attribute__((ext_vector_type(8)))  int      i32x8;
typedef __attribute__((ext_vector_type(4)))  int      i32x4;

// ---------------------------------------------------------------------------
// WMMA fragment helpers (CDNA5 wave32 layouts, CDNA5 ISA §7.12.2)
// ---------------------------------------------------------------------------
__device__ __forceinline__ v16h frag_a_f32(const float* __restrict__ src, int lda,
                                           int m0, int k0, int lane) {
  const int half = lane >> 4, m = lane & 15;
  const float* row = src + (size_t)(m0 + m) * lda + k0;
  v16h a;
#pragma unroll
  for (int i = 0; i < 8; ++i) {
    const int kb = ((i < 4) ? (2 * i) : (16 + 2 * (i - 4))) + 8 * half;
    a[2 * i]     = (_Float16)row[kb];
    a[2 * i + 1] = (_Float16)row[kb + 1];
  }
  return a;
}

// B (32x16 f16): lanes 0-15 -> N=0..15 with K=0..7,16..23, lanes 16-31 -> K=8..15,24..31.
__device__ __forceinline__ v16h frag_b_f32(const float* __restrict__ src, int ldb,
                                           int k0, int n0, int lane) {
  const int half = lane >> 4, n = lane & 15;
  const float* col = src + n0 + n;
  v16h b;
#pragma unroll
  for (int i = 0; i < 8; ++i) {
    const int k = k0 + ((i < 4) ? (2 * i) : (16 + 2 * (i - 4))) + 8 * half;
    b[2 * i]     = (_Float16)col[(size_t)k * ldb];
    b[2 * i + 1] = (_Float16)col[(size_t)(k + 1) * ldb];
  }
  return b;
}

// C/D (16x16 f32): VGPR r, lanes 0-15 -> M=r, N=lane; lanes 16-31 -> M=r+8.
__device__ __forceinline__ void store_c(float* __restrict__ dst, int ldc,
                                        int m0, int n0, int lane, v8f c,
                                        const float* __restrict__ bias) {
  const int half = lane >> 4, n = lane & 15;
  const float bn = bias[n0 + n];
#pragma unroll
  for (int r = 0; r < 8; ++r) {
    const int m = r + 8 * half;
    dst[(size_t)(m0 + m) * ldc + n0 + n] = c[r] + bn;
  }
}

// ---------------------------------------------------------------------------
// Kernel 0: fold output matmuls: Wc = Wo(192x64) @ Wf(64x64), bc = bo@Wf + bf
// (no nonlinearity between them in the reference, so folding is exact).
// ---------------------------------------------------------------------------
__global__ __launch_bounds__(256) void fold_weights(
    const float* __restrict__ Wo, const float* __restrict__ bo,
    const float* __restrict__ Wf, const float* __restrict__ bf,
    float* __restrict__ Wc, float* __restrict__ bc) {
  const int t = blockIdx.x * 256 + threadIdx.x;
  if (t < CAT * PROP) {
    const int r = t >> 6, c = t & 63;
    float acc = 0.f;
#pragma unroll 8
    for (int k = 0; k < PROP; ++k) acc += Wo[r * PROP + k] * Wf[k * PROP + c];
    Wc[t] = acc;
  }
  if (t < PROP) {
    float acc = bf[t];
#pragma unroll 8
    for (int k = 0; k < PROP; ++k) acc += bo[k] * Wf[k * PROP + t];
    bc[t] = acc;
  }
}

// ---------------------------------------------------------------------------
// Kernel 1: pre-pack weights to f16 in WMMA B-fragment lane layout:
// pW[frag*32 + lane] = 16 halves that lane feeds into v_wmma. GEMM kernels
// then load each fragment as two coalesced b128s instead of 16 strided b32s.
// Wh: 2 ksteps x 4 ntiles = 8 frags. Wc: 6 x 4 = 24 frags. 1024 threads.
// ---------------------------------------------------------------------------
__global__ __launch_bounds__(256) void pack_weights(
    const float* __restrict__ Wh, const float* __restrict__ Wc,
    v16h* __restrict__ pWh, v16h* __restrict__ pWc) {
  const int t = blockIdx.x * 256 + threadIdx.x;
  const int lane = t & 31;
  const int frag = t >> 5;
  if (frag < 8) {
    const int ks = frag >> 2, nt = frag & 3;
    pWh[frag * 32 + lane] = frag_b_f32(Wh, PROP, ks * 32, nt * 16, lane);
  } else {
    const int f = frag - 8;                 // 0..23
    const int ks = f >> 2, nt = f & 3;
    pWc[f * 32 + lane] = frag_b_f32(Wc, PROP, ks * 32, nt * 16, lane);
  }
}

// ---------------------------------------------------------------------------
// Kernel 2: coords = x @ Ws + bs  (S=4, pure VALU fp32 — too thin for WMMA)
// ---------------------------------------------------------------------------
__global__ __launch_bounds__(256) void coords_kernel(
    const float* __restrict__ x, const float* __restrict__ Ws,
    const float* __restrict__ bs, float* __restrict__ coords) {
  const int n = blockIdx.x * 256 + threadIdx.x;   // exact grid: BN threads
  const float* xr = x + (size_t)n * IN_DIM;
  float a0 = bs[0], a1 = bs[1], a2 = bs[2], a3 = bs[3];
#pragma unroll 8
  for (int k = 0; k < IN_DIM; ++k) {
    const float v = xr[k];
    a0 += v * Ws[k * S_DIM + 0];
    a1 += v * Ws[k * S_DIM + 1];
    a2 += v * Ws[k * S_DIM + 2];
    a3 += v * Ws[k * S_DIM + 3];
  }
  float* cr = coords + (size_t)n * S_DIM;
  cr[0] = a0; cr[1] = a1; cr[2] = a2; cr[3] = a3;
}

// ---------------------------------------------------------------------------
// Kernel 3: feats = x @ Wh + bh via v_wmma_f32_16x16x32_f16, packed B.
// One wave = 16 rows x 64 cols (2 K-steps x 4 N-tiles = 8 WMMAs). Exact grid.
// ---------------------------------------------------------------------------
__global__ __launch_bounds__(256) void feats_kernel(
    const float* __restrict__ x, const v16h* __restrict__ pWh,
    const float* __restrict__ bh, float* __restrict__ feats) {
  const int lane = threadIdx.x & 31;
  const int wave = (blockIdx.x * 256 + threadIdx.x) >> 5;
  const int row0 = wave * 16;

  const v16h a0 = frag_a_f32(x, IN_DIM, row0, 0, lane);
  const v16h a1 = frag_a_f32(x, IN_DIM, row0, 32, lane);
#pragma unroll
  for (int nt = 0; nt < 4; ++nt) {
    const v16h b0 = pWh[(0 * 4 + nt) * 32 + lane];
    const v16h b1 = pWh[(1 * 4 + nt) * 32 + lane];
    v8f c = {};
    c = __builtin_amdgcn_wmma_f32_16x16x32_f16(false, a0, false, b0, (short)0, c, false, false);
    c = __builtin_amdgcn_wmma_f32_16x16x32_f16(false, a1, false, b1, (short)0, c, false, false);
    store_c(feats, PROP, row0, nt * 16, lane, c, bh);
  }
}

// ---------------------------------------------------------------------------
// Kernel 4: per-node kNN (top-40 smallest d2) within each 2048-node segment.
// Segment coords (32 KB) are DMA'd into LDS with the Tensor Data Mover when
// available (tensor_load_to_lds + s_wait_tensorcnt), else cooperative b128s.
// Distance scan reads one float4 LDS broadcast per candidate. Top-K kept as
// packed keys in LDS: [31:11]=fp32 d2 high bits (d2>=0 -> uint order == float
// order), [10:0]=index. 128 thr * 40 * 4B = 20 KB.
// ---------------------------------------------------------------------------
__global__ __launch_bounds__(128) void knn_kernel(
    const float* __restrict__ coords, int* __restrict__ idx_out,
    float* __restrict__ w_out) {
  __shared__ float4   sc4[N_SEG];          // 32 KB
  __shared__ unsigned sk[128 * KNN];       // 20 KB

  const int seg  = blockIdx.y;
  const int node = blockIdx.x * 128 + threadIdx.x;
  const float* segc = coords + (size_t)seg * N_SEG * S_DIM;

#if __has_builtin(__builtin_amdgcn_tensor_load_to_lds) && __has_builtin(__builtin_amdgcn_s_wait_tensorcnt)
  // TDM bulk copy: 1-D tensor, 8192 fp32 elements, whole tile in one D#.
  if (threadIdx.x == 0) {
    const unsigned lds_off = (unsigned)(uintptr_t)(&sc4[0]);
    const unsigned long long ga = (unsigned long long)(uintptr_t)segc;
    u32x4 g0;
    g0[0] = 1u;                                            // count=1 valid D#
    g0[1] = lds_off;                                       // lds_addr
    g0[2] = (unsigned)(ga & 0xFFFFFFFFu);                  // global_addr lo
    g0[3] = (unsigned)((ga >> 32) & 0x01FFFFFFu) | (2u << 30); // addr hi | type=2
    i32x8 g1;
    g1[0] = (int)(2u << 16);        // data_size = 4 bytes
    g1[1] = (int)(0x2000u << 16);   // tensor_dim0 = 8192 (low 16 bits)
    g1[2] = (int)0x00010000;        // tensor_dim1 = 1
    g1[3] = (int)(0x2000u << 16);   // tile_dim0 = 8192
    g1[4] = 1;                      // tile_dim1 = 1
    g1[5] = 8192;                   // tensor_dim0_stride
    g1[6] = 0; g1[7] = 0;
    i32x4 z4; z4[0] = z4[1] = z4[2] = z4[3] = 0;
#if __has_include(<hip/amd_detail/amd_gfx1250_TDM.h>)
    i32x8 z8; z8[0]=z8[1]=z8[2]=z8[3]=z8[4]=z8[5]=z8[6]=z8[7]=0;
    __builtin_amdgcn_tensor_load_to_lds(g0, g1, z4, z4, z8, 0);
#else
    __builtin_amdgcn_tensor_load_to_lds(g0, g1, z4, z4, 0);
#endif
    __builtin_amdgcn_s_wait_tensorcnt(0);
  }
  __syncthreads();
#else
  {
    const float4* segc4 = (const float4*)segc;
    for (int i = threadIdx.x; i < N_SEG; i += 128) sc4[i] = segc4[i];
    __syncthreads();
  }
#endif

  const float4 cm = sc4[node];
  unsigned* kl = sk + threadIdx.x * KNN;
  int cnt = 0;
#pragma unroll 2
  for (int j = 0; j < N_SEG; ++j) {
    const float4 cj = sc4[j];                 // same j all lanes: LDS broadcast
    const float d0 = cj.x - cm.x, d1 = cj.y - cm.y;
    const float d2 = cj.z - cm.z, d3 = cj.w - cm.w;
    const float dd = d0 * d0 + d1 * d1 + d2 * d2 + d3 * d3;
    const unsigned key = (__float_as_uint(dd) & 0xFFFFF800u) | (unsigned)j;
    if (cnt < KNN) {
      int p = cnt++;
      while (p > 0 && kl[p - 1] > key) { kl[p] = kl[p - 1]; --p; }
      kl[p] = key;
    } else if (key < kl[KNN - 1]) {
      int p = KNN - 1;
      while (p > 0 && kl[p - 1] > key) { kl[p] = kl[p - 1]; --p; }
      kl[p] = key;
    }
  }

  const size_t g = (size_t)seg * N_SEG + node;
#pragma unroll 8
  for (int k = 0; k < KNN; ++k) {
    const unsigned key = kl[k];
    float dd = __uint_as_float(key & 0xFFFFF800u);
    dd = dd < 0.f ? 0.f : dd;
    w_out[g * KNN + k]   = __expf(-10.f * dd);   // v_exp_f32
    idx_out[g * KNN + k] = (int)(key & 0x7FFu);
  }
}

// ---------------------------------------------------------------------------
// Kernel 5: weighted gather + mean/max, builds concat = [x | mean | max].
// One wave per node; neighbor index is wave-uniform -> broadcast loads.
// ---------------------------------------------------------------------------
__global__ __launch_bounds__(256) void agg_kernel(
    const float* __restrict__ x, const float* __restrict__ feats,
    const int* __restrict__ idx, const float* __restrict__ w,
    float* __restrict__ concat) {
  const int lane = threadIdx.x & 31;
  const int n    = (blockIdx.x * 256 + threadIdx.x) >> 5;   // exact: BN waves
  const int seg_base = (n >> 11) << 11;

  const int*   ip = idx + (size_t)n * KNN;
  const float* wp = w   + (size_t)n * KNN;

  float m0 = 0.f, m1 = 0.f;
  float x0 = -3.0e38f, x1 = -3.0e38f;
#pragma unroll 4
  for (int k = 0; k < KNN; ++k) {
    const int j = seg_base + ip[k];
    if (k + 1 < KNN)
      __builtin_prefetch(&feats[(size_t)(seg_base + ip[k + 1]) * PROP], 0, 0);
    const float wk = wp[k];
    const float v0 = wk * feats[(size_t)j * PROP + lane];
    const float v1 = wk * feats[(size_t)j * PROP + lane + 32];
    m0 += v0; m1 += v1;
    x0 = fmaxf(x0, v0); x1 = fmaxf(x1, v1);
  }
  const float* xr = x + (size_t)n * IN_DIM;
  float* cp = concat + (size_t)n * CAT;
  cp[lane]             = xr[lane];
  cp[lane + 32]        = xr[lane + 32];
  cp[IN_DIM + lane]      = m0 * (1.f / KNN);
  cp[IN_DIM + lane + 32] = m1 * (1.f / KNN);
  cp[IN_DIM + PROP + lane]      = x0;
  cp[IN_DIM + PROP + lane + 32] = x1;
}

// ---------------------------------------------------------------------------
// Kernel 6: out = concat(16384x192) @ Wc(192x64) + bc, packed B.
// 6 K-steps x 4 N-tiles = 24 WMMAs per wave.
// ---------------------------------------------------------------------------
__global__ __launch_bounds__(256) void out_gemm(
    const float* __restrict__ concat, const v16h* __restrict__ pWc,
    const float* __restrict__ bc, float* __restrict__ out) {
  const int lane = threadIdx.x & 31;
  const int wave = (blockIdx.x * 256 + threadIdx.x) >> 5;
  const int row0 = wave * 16;

  v8f acc[4] = {v8f{}, v8f{}, v8f{}, v8f{}};
#pragma unroll
  for (int ks = 0; ks < CAT / 32; ++ks) {
    const v16h a = frag_a_f32(concat, CAT, row0, ks * 32, lane);
#pragma unroll
    for (int nt = 0; nt < 4; ++nt) {
      const v16h b = pWc[(ks * 4 + nt) * 32 + lane];
      acc[nt] = __builtin_amdgcn_wmma_f32_16x16x32_f16(false, a, false, b,
                                                       (short)0, acc[nt], false, false);
    }
  }
#pragma unroll
  for (int nt = 0; nt < 4; ++nt)
    store_c(out, PROP, row0, nt * 16, lane, acc[nt], bc);
}

// ---------------------------------------------------------------------------
// Launch. Inputs (setup_inputs order):
// 0:x 1:row_splits 2:Ws 3:bs 4:Wh 5:bh 6:Wo 7:bo 8:Wf 9:bf
// ---------------------------------------------------------------------------
extern "C" void kernel_launch(void* const* d_in, const int* in_sizes, int n_in,
                              void* d_out, int out_size, void* d_ws, size_t ws_size,
                              hipStream_t stream) {
  const float* x  = (const float*)d_in[0];
  const float* Ws = (const float*)d_in[2];
  const float* bs = (const float*)d_in[3];
  const float* Wh = (const float*)d_in[4];
  const float* bh = (const float*)d_in[5];
  const float* Wo = (const float*)d_in[6];
  const float* bo = (const float*)d_in[7];
  const float* Wf = (const float*)d_in[8];
  const float* bf = (const float*)d_in[9];
  float* out = (float*)d_out;

  // Workspace layout; every block is a multiple of 256B so v16h stays aligned.
  char* p = (char*)d_ws;
  size_t o = 0;
  float* coords  = (float*)(p + o); o += (size_t)BN * S_DIM * 4;      // 256 KB
  float* feats   = (float*)(p + o); o += (size_t)BN * PROP * 4;       // 4 MB
  float* Wc      = (float*)(p + o); o += (size_t)CAT * PROP * 4;      // 48 KB
  float* bc      = (float*)(p + o); o += (size_t)PROP * 4;            // 256 B
  v16h*  pWh     = (v16h*)(p + o);  o += (size_t)8 * 32 * sizeof(v16h);  // 8 KB
  v16h*  pWc     = (v16h*)(p + o);  o += (size_t)24 * 32 * sizeof(v16h); // 24 KB
  int*   knn_idx = (int*)(p + o);   o += (size_t)BN * KNN * 4;        // 2.5 MB
  float* knn_w   = (float*)(p + o); o += (size_t)BN * KNN * 4;        // 2.5 MB
  float* concat  = (float*)(p + o);                                    // 12 MB

  fold_weights<<<(CAT * PROP + 255) / 256, 256, 0, stream>>>(Wo, bo, Wf, bf, Wc, bc);
  pack_weights<<<4, 256, 0, stream>>>(Wh, Wc, pWh, pWc);
  coords_kernel<<<BN / 256, 256, 0, stream>>>(x, Ws, bs, coords);
  feats_kernel<<<(BN / 16) / 8, 256, 0, stream>>>(x, pWh, bh, feats);      // 1024 waves
  knn_kernel<<<dim3(N_SEG / 128, B_SEG), 128, 0, stream>>>(coords, knn_idx, knn_w);
  agg_kernel<<<BN / 8, 256, 0, stream>>>(x, feats, knn_idx, knn_w, concat); // BN waves
  out_gemm<<<(BN / 16) / 8, 256, 0, stream>>>(concat, pWc, bc, out);        // 1024 waves
}
